// JetBlockAttentionV4_60163901882948
// MI455X (gfx1250) — matrix operand order repaired and checked by reading, not compiled, not run on hardware
//
#include <hip/hip_runtime.h>

// ---------------------------------------------------------------- constants
#define B_      2
#define S_      2048
#define H_      1024
#define NH_     16
#define NKV_    4
#define HD_     64
#define SPLIT_  64
#define NCHUNK_ 32          // S_/SPLIT_
#define M_      (B_*S_)     // 4096 rows

// ---------------------------------------------------------------- types
typedef __bf16 bf16_t;
typedef bf16_t v8bf  __attribute__((ext_vector_type(8)));
typedef bf16_t v16bf __attribute__((ext_vector_type(16)));
typedef float  v8f   __attribute__((ext_vector_type(8)));
typedef unsigned int u32x4 __attribute__((ext_vector_type(4)));
typedef int i32x8 __attribute__((ext_vector_type(8)));
typedef int i32x4 __attribute__((ext_vector_type(4)));

#if defined(__has_builtin)
#if __has_builtin(__builtin_amdgcn_tensor_load_to_lds) && __has_builtin(__builtin_amdgcn_s_wait_tensorcnt)
#define USE_TDM 1
#endif
#if __has_builtin(__builtin_amdgcn_global_load_async_to_lds_b128) && __has_builtin(__builtin_amdgcn_s_wait_asynccnt)
#define USE_ASYNC 1
#endif
#endif
#ifndef USE_TDM
#define USE_TDM 0
#endif
#ifndef USE_ASYNC
#define USE_ASYNC 0
#endif

// ---------------------------------------------------------------- helpers
static __device__ __forceinline__ unsigned short f2bf(float f) {
  unsigned u = __builtin_bit_cast(unsigned, f);
  u += 0x7fffu + ((u >> 16) & 1u);          // round-to-nearest-even
  return (unsigned short)(u >> 16);
}

#if USE_ASYNC
// GLOBAL_LOAD_ASYNC_TO_LDS_B128: 16B per lane, memory -> LDS, ASYNCcnt-tracked.
// Builtin signature (from hipcc diagnostic): AS1 ptr to <4 x i32>, AS3 ptr, imm, imm.
typedef int v4i_vs __attribute__((vector_size(16)));
typedef __attribute__((address_space(1))) v4i_vs* as1_v4i_p;
typedef __attribute__((address_space(3))) v4i_vs* as3_v4i_p;
static __device__ __forceinline__ void async_cp_b128(const void* g, void* l) {
  as1_v4i_p gp = (as1_v4i_p)(unsigned long long)(size_t)g;
  as3_v4i_p lp = (as3_v4i_p)(unsigned)(size_t)l;
  __builtin_amdgcn_global_load_async_to_lds_b128(gp, lp, 0, 0);
}
#endif

// A-fragment (16x32 bf16, MxK): lanes 0-15 hold K=[kb..kb+7],[kb+16..kb+23];
// lanes 16-31 hold K=[kb+8..kb+15],[kb+24..kb+31]  (ISA 7.12.2)
static __device__ __forceinline__ v16bf frag_a(const unsigned short* rowptr, int kb, int lane) {
  const int lo = kb + ((lane & 16) ? 8 : 0);
  v8bf a0 = *(const v8bf*)(rowptr + lo);
  v8bf a1 = *(const v8bf*)(rowptr + lo + 16);
  return __builtin_shufflevector(a0, a1, 0,1,2,3,4,5,6,7,8,9,10,11,12,13,14,15);
}
// B-fragment (32x16 bf16, KxN): lane = column; lanes 0-15 hold K=[kb..kb+7],[kb+16..kb+23];
// lanes 16-31 hold K=[kb+8..kb+15],[kb+24..kb+31] (same as A).
static __device__ __forceinline__ v16bf frag_b(const unsigned short* rowptr, int kb, int lane) {
  const int lo = kb + ((lane & 16) ? 8 : 0);
  v8bf b0 = *(const v8bf*)(rowptr + lo);
  v8bf b1 = *(const v8bf*)(rowptr + lo + 16);
  return __builtin_shufflevector(b0, b1, 0,1,2,3,4,5,6,7,8,9,10,11,12,13,14,15);
}
static __device__ __forceinline__ v8f wmma_bf16(v16bf a, v16bf b, v8f c) {
  return __builtin_amdgcn_wmma_f32_16x16x32_bf16(false, a, false, b, (short)0, c, false, false);
}

// ---------------------------------------------------------------- 1) gate
__global__ __launch_bounds__(256) void gate_kernel(
    const float* __restrict__ x, const float* __restrict__ w_kgen,
    const float* __restrict__ b_kgen, float* __restrict__ gate)
{
  const int b = blockIdx.x, tid = threadIdx.x;
  float p0 = 0.f, p1 = 0.f, p2 = 0.f;
  for (int h = tid; h < H_; h += 256) {
    const float* col = x + (size_t)b * S_ * H_ + h;
    float s = 0.f;
    for (int t = 0; t < S_; ++t) s += col[(size_t)t * H_];
    const float ctx = s * (1.0f / (float)S_);
    p0 += ctx * w_kgen[0 * H_ + h];
    p1 += ctx * w_kgen[1 * H_ + h];
    p2 += ctx * w_kgen[2 * H_ + h];
  }
  __shared__ float red[3][256];
  red[0][tid] = p0; red[1][tid] = p1; red[2][tid] = p2;
  __syncthreads();
  for (int off = 128; off > 0; off >>= 1) {
    if (tid < off)
      for (int j = 0; j < 3; ++j) red[j][tid] += red[j][tid + off];
    __syncthreads();
  }
  if (tid == 0) {
    float g = 0.f;
    for (int j = 0; j < 3; ++j)
      g += 1.0f / (1.0f + __expf(-(red[j][0] + b_kgen[j])));
    gate[b] = g * (1.0f / 3.0f);
  }
}

// ---------------------------------------------------------------- 2) conv -> bf16
__global__ __launch_bounds__(256) void dynconv_kernel(
    const float* __restrict__ x, const float* __restrict__ w_conv,
    const float* __restrict__ b_conv, const float* __restrict__ gate,
    unsigned short* __restrict__ h)
{
  const int idx = blockIdx.x * 256 + threadIdx.x;      // (b*S+s)*H + c
  const int c = idx & (H_ - 1);
  const int s = (idx >> 10) & (S_ - 1);
  const int b = idx >> 21;
  const size_t base = (size_t)idx;
  float acc = b_conv[c];
  const float w0 = w_conv[c * 3 + 0], w1 = w_conv[c * 3 + 1], w2 = w_conv[c * 3 + 2];
  if (s > 0)      acc += w0 * x[base - H_];
  acc += w1 * x[base];
  if (s < S_ - 1) acc += w2 * x[base + H_];
  h[idx] = f2bf(acc * gate[b]);
}

// ---------------------------------------------------------------- 3) f32 -> bf16
__global__ __launch_bounds__(256) void cvt_bf16(
    const float* __restrict__ src, unsigned short* __restrict__ dst)
{
  const int i = blockIdx.x * 256 + threadIdx.x;
  dst[i] = f2bf(src[i]);
}

// ---------------------------------------------------------------- 4) WMMA GEMM
// Y[M,N] = A[M,K](bf16) * W[N,K]^T(bf16).  Block tile 128x64, 8 waves in a
// 4x2 grid; each wave owns a 32x32 register tile (2x2 WMMA tiles, 4 v8f accs)
// so A/B fragments are each reused twice -> 4 WMMAs per 8 ds_load_b128.
template<int OUT_F32>
__global__ __launch_bounds__(256) void gemm_bf16_wmma(
    const unsigned short* __restrict__ A,
    const unsigned short* __restrict__ W,
    void* __restrict__ Yv, int M, int N, int K)
{
  __shared__ unsigned short As[128][32];   // 8 KB
  __shared__ unsigned short Bs[64][32];    // 4 KB
  const int tid = threadIdx.x, lane = tid & 31, wid = tid >> 5;
  const int m0 = blockIdx.x * 128, n0 = blockIdx.y * 64;
  const int rb  = (wid >> 1) * 32;         // wave row base (0,32,64,96)
  const int cbk = (wid & 1) * 32;          // wave col base (0,32)
  v8f acc00 = {}, acc01 = {}, acc10 = {}, acc11 = {};

  for (int k0 = 0; k0 < K; k0 += 32) {
    __syncthreads();
    { // stage A: 128x32 bf16, 32B per thread
      const int r = tid >> 1, c = (tid & 1) * 16;
      const uint4* src = (const uint4*)(A + (size_t)(m0 + r) * K + k0 + c);
      uint4* dst = (uint4*)&As[r][c];
      dst[0] = src[0]; dst[1] = src[1];
      if (k0 + 32 < K)
        __builtin_prefetch(A + (size_t)(m0 + r) * K + k0 + 32 + c, 0, 1);
    }
#if USE_TDM
    if (wid == 7) {
      // Tensor Data Mover: 64x32-element bf16 tile of W -> Bs (TENSORcnt)
      const unsigned long long ga =
          (unsigned long long)(size_t)(const void*)(W + (size_t)n0 * K + k0);
      const unsigned lds = (unsigned)(size_t)(void*)&Bs[0][0];
      u32x4 g0;
      g0[0] = 1u;                                          // count=1, user mode
      g0[1] = lds;                                         // lds_addr
      g0[2] = (unsigned)ga;                                // global_addr[31:0]
      g0[3] = (unsigned)((ga >> 32) & 0x01ffffffu) | (2u << 30); // addr[56:32], type=2
      i32x8 g1;
      g1[0] = (int)(1u << 16);                             // data_size = 2B
      g1[1] = (int)(((unsigned)K & 0xffffu) << 16);        // tensor_dim0[15:0]
      g1[2] = (int)(((unsigned)K >> 16) | (64u << 16));    // dim0 hi | tensor_dim1=64
      g1[3] = (int)(32u << 16);                            // tile_dim0 = 32
      g1[4] = 64;                                          // tile_dim1 = 64
      g1[5] = K;                                           // tensor_dim0_stride
      g1[6] = 0; g1[7] = 0;
      i32x4 z4 = {0, 0, 0, 0};
      i32x8 z8 = {0, 0, 0, 0, 0, 0, 0, 0};
      __builtin_amdgcn_tensor_load_to_lds(g0, g1, z4, z4, z8, 0);
      __builtin_amdgcn_s_wait_tensorcnt(0);
    }
#else
    if (tid < 128) {                       // stage B: 64x32 bf16, 32B per thread
      const int r = tid >> 1, c = (tid & 1) * 16;
      const uint4* src = (const uint4*)(W + (size_t)(n0 + r) * K + k0 + c);
      uint4* dst = (uint4*)&Bs[r][c];
      dst[0] = src[0]; dst[1] = src[1];
    }
#endif
    __syncthreads();
    const int ln = lane & 15;
    v16bf a0 = frag_a(&As[rb + ln][0], 0, lane);
    v16bf a1 = frag_a(&As[rb + 16 + ln][0], 0, lane);
    v16bf b0 = frag_b(&Bs[cbk + ln][0], 0, lane);
    v16bf b1 = frag_b(&Bs[cbk + 16 + ln][0], 0, lane);
    acc00 = wmma_bf16(a0, b0, acc00);
    acc01 = wmma_bf16(a0, b1, acc01);
    acc10 = wmma_bf16(a1, b0, acc10);
    acc11 = wmma_bf16(a1, b1, acc11);
  }

  const int nc = lane & 15;
  const int mh = (lane & 16) ? 8 : 0;
  for (int j = 0; j < 8; ++j) {
    const int ma = m0 + rb + mh + j;
    const int mb = ma + 16;
    const int na = n0 + cbk + nc;
    if (OUT_F32) {
      float* Y = (float*)Yv;
      Y[(size_t)ma * N + na]      = acc00[j];
      Y[(size_t)ma * N + na + 16] = acc01[j];
      Y[(size_t)mb * N + na]      = acc10[j];
      Y[(size_t)mb * N + na + 16] = acc11[j];
    } else {
      unsigned short* Y = (unsigned short*)Yv;
      Y[(size_t)ma * N + na]      = f2bf(acc00[j]);
      Y[(size_t)ma * N + na + 16] = f2bf(acc01[j]);
      Y[(size_t)mb * N + na]      = f2bf(acc10[j]);
      Y[(size_t)mb * N + na + 16] = f2bf(acc11[j]);
    }
  }
}

// ---------------------------------------------------------------- 5) chunked attention
// One block per (qtile=64, head, batch). Chunk-local softmax; since each softmax
// row sums to 1, the reference combine == mean over chunks -> accumulate & /32.
__global__ __launch_bounds__(256) void attn_chunked_wmma(
    const unsigned short* __restrict__ Q,    // (B*S) x 1024 bf16
    const unsigned short* __restrict__ Kg,   // (B*S) x 256  bf16
    const unsigned short* __restrict__ Vg,   // (B*S) x 256  bf16
    unsigned short* __restrict__ O)          // (B*S) x 1024 bf16
{
  __shared__ unsigned short Qs[64][64];
  __shared__ unsigned short Ks[64][64];     // [token][d]
  __shared__ unsigned short Vs[64][64];     // transposed: [d][token]
  __shared__ float          Sf[64][64];
  __shared__ unsigned short Ps[64][64];

  const int tid = threadIdx.x, lane = tid & 31, wid = tid >> 5;
  const int q0   = blockIdx.x * 64;
  const int head = blockIdx.y;
  const int b    = blockIdx.z;
  const int kvh  = head >> 2;               // GQA: 4 q-heads per kv-head
  const int rt = wid >> 1;                  // 16-row tile 0..3
  const int cb = (wid & 1) * 2;             // 16-col tile base (0 or 2)

  { // stage Q tile once (async -> LDS when available)
    const int r = tid >> 2, c = (tid & 3) * 16;
    const unsigned short* src = Q + (size_t)(b * S_ + q0 + r) * (NH_ * HD_) + head * HD_ + c;
#if USE_ASYNC
    async_cp_b128(src,     &Qs[r][c]);
    async_cp_b128(src + 8, &Qs[r][c + 8]);
#else
    uint4* dst = (uint4*)&Qs[r][c];
    dst[0] = ((const uint4*)src)[0]; dst[1] = ((const uint4*)src)[1];
#endif
  }

  v8f o0 = {}, o1 = {};
  const float scale = 0.125f;               // HD^-0.5

  for (int ch = 0; ch < NCHUNK_; ++ch) {
    __syncthreads();                        // prev PV-WMMA done reading Ks/Vs
    const int t0 = ch * SPLIT_;
    { // stage K tile (async) + V tile (transposed, via registers)
      const int r = tid >> 2, c = (tid & 3) * 16;
      const unsigned short* sk = Kg + (size_t)(b * S_ + t0 + r) * (NKV_ * HD_) + kvh * HD_ + c;
#if USE_ASYNC
      async_cp_b128(sk,     &Ks[r][c]);
      async_cp_b128(sk + 8, &Ks[r][c + 8]);
#else
      uint4* dk = (uint4*)&Ks[r][c];
      dk[0] = ((const uint4*)sk)[0]; dk[1] = ((const uint4*)sk)[1];
#endif
      __attribute__((aligned(16))) unsigned short tmp[16];
      const uint4* sv = (const uint4*)(Vg + (size_t)(b * S_ + t0 + r) * (NKV_ * HD_) + kvh * HD_ + c);
      *(uint4*)&tmp[0] = sv[0]; *(uint4*)&tmp[8] = sv[1];
      for (int j = 0; j < 16; ++j) Vs[c + j][r] = tmp[j];
#if USE_ASYNC
      __builtin_amdgcn_s_wait_asynccnt(0);  // async K (and Q on iter 0) landed
#endif
    }
    __syncthreads();

    // scores S = Q * Kc^T  (K-dim = HD = 2 WMMA steps)
    v8f s0v = {}, s1v = {};
    const unsigned short* qrow  = &Qs[rt * 16 + (lane & 15)][0];
    const unsigned short* k0row = &Ks[(cb + 0) * 16 + (lane & 15)][0];
    const unsigned short* k1row = &Ks[(cb + 1) * 16 + (lane & 15)][0];
    for (int db = 0; db < HD_; db += 32) {
      v16bf a = frag_a(qrow, db, lane);
      s0v = wmma_bf16(a, frag_b(k0row, db, lane), s0v);
      s1v = wmma_bf16(a, frag_b(k1row, db, lane), s1v);
    }
    {
      const int nc = lane & 15, mh = (lane & 16) ? 8 : 0;
      for (int j = 0; j < 8; ++j) {
        Sf[rt * 16 + mh + j][(cb + 0) * 16 + nc] = s0v[j] * scale;
        Sf[rt * 16 + mh + j][(cb + 1) * 16 + nc] = s1v[j] * scale;
      }
    }
    __syncthreads();

    // chunk-local softmax: one row (64 keys) per thread
    if (tid < 64) {
      float mx = -1e30f;
      for (int j = 0; j < 64; ++j) mx = fmaxf(mx, Sf[tid][j]);
      float sum = 0.f;
      for (int j = 0; j < 64; ++j) sum += __expf(Sf[tid][j] - mx);
      const float inv = 1.0f / sum;
      for (int j = 0; j < 64; ++j) Ps[tid][j] = f2bf(__expf(Sf[tid][j] - mx) * inv);
    }
    __syncthreads();

    // O += P * Vc  (K-dim = SPLIT = 2 WMMA steps), accumulate across chunks
    const unsigned short* prow  = &Ps[rt * 16 + (lane & 15)][0];
    const unsigned short* v0row = &Vs[(cb + 0) * 16 + (lane & 15)][0];
    const unsigned short* v1row = &Vs[(cb + 1) * 16 + (lane & 15)][0];
    for (int kb = 0; kb < SPLIT_; kb += 32) {
      v16bf a = frag_a(prow, kb, lane);
      o0 = wmma_bf16(a, frag_b(v0row, kb, lane), o0);
      o1 = wmma_bf16(a, frag_b(v1row, kb, lane), o1);
    }
  }

  const float invC = 1.0f / (float)NCHUNK_;   // sum_w == 1 per chunk -> mean
  const int nc = lane & 15, mh = (lane & 16) ? 8 : 0;
  for (int j = 0; j < 8; ++j) {
    const int m  = b * S_ + q0 + rt * 16 + mh + j;
    const int c0 = head * HD_ + (cb + 0) * 16 + nc;
    O[(size_t)m * (NH_ * HD_) + c0]      = f2bf(o0[j] * invC);
    O[(size_t)m * (NH_ * HD_) + c0 + 16] = f2bf(o1[j] * invC);
  }
}

// ---------------------------------------------------------------- launch
extern "C" void kernel_launch(void* const* d_in, const int* in_sizes, int n_in,
                              void* d_out, int out_size, void* d_ws, size_t ws_size,
                              hipStream_t stream) {
  const float* x      = (const float*)d_in[0];
  const float* w_q    = (const float*)d_in[1];
  const float* w_k    = (const float*)d_in[2];
  const float* w_v    = (const float*)d_in[3];
  const float* w_o    = (const float*)d_in[4];
  const float* w_kgen = (const float*)d_in[5];
  const float* b_kgen = (const float*)d_in[6];
  const float* w_conv = (const float*)d_in[7];
  const float* b_conv = (const float*)d_in[8];

  char* ws = (char*)d_ws;
  size_t off = 0;
  float* gate = (float*)(ws + off);            off += 256;
  unsigned short* h_bf = (unsigned short*)(ws + off); off += (size_t)M_ * H_ * 2;      // 8 MB
  unsigned short* wq   = (unsigned short*)(ws + off); off += (size_t)H_ * H_ * 2;      // 2 MB
  unsigned short* wk   = (unsigned short*)(ws + off); off += (size_t)(NKV_*HD_) * H_ * 2;
  unsigned short* wv   = (unsigned short*)(ws + off); off += (size_t)(NKV_*HD_) * H_ * 2;
  unsigned short* wo   = (unsigned short*)(ws + off); off += (size_t)H_ * H_ * 2;
  unsigned short* qb   = (unsigned short*)(ws + off); off += (size_t)M_ * (NH_*HD_) * 2;
  unsigned short* kbuf = (unsigned short*)(ws + off); off += (size_t)M_ * (NKV_*HD_) * 2;
  unsigned short* vbuf = (unsigned short*)(ws + off); off += (size_t)M_ * (NKV_*HD_) * 2;
  unsigned short* ab   = (unsigned short*)(ws + off); off += (size_t)M_ * (NH_*HD_) * 2;

  // 1) gate
  gate_kernel<<<B_, 256, 0, stream>>>(x, w_kgen, b_kgen, gate);
  // 2) depthwise conv * gate -> h (bf16)
  dynconv_kernel<<<(M_ * H_) / 256, 256, 0, stream>>>(x, w_conv, b_conv, gate, h_bf);
  // 3) weights -> bf16
  cvt_bf16<<<(H_ * H_) / 256, 256, 0, stream>>>(w_q, wq);
  cvt_bf16<<<(NKV_ * HD_ * H_) / 256, 256, 0, stream>>>(w_k, wk);
  cvt_bf16<<<(NKV_ * HD_ * H_) / 256, 256, 0, stream>>>(w_v, wv);
  cvt_bf16<<<(H_ * H_) / 256, 256, 0, stream>>>(w_o, wo);
  // 4) projections (bf16 WMMA, f32 accum)
  gemm_bf16_wmma<0><<<dim3(M_ / 128, (NH_ * HD_) / 64), 256, 0, stream>>>(h_bf, wq, qb,   M_, NH_ * HD_,  H_);
  gemm_bf16_wmma<0><<<dim3(M_ / 128, (NKV_ * HD_) / 64), 256, 0, stream>>>(h_bf, wk, kbuf, M_, NKV_ * HD_, H_);
  gemm_bf16_wmma<0><<<dim3(M_ / 128, (NKV_ * HD_) / 64), 256, 0, stream>>>(h_bf, wv, vbuf, M_, NKV_ * HD_, H_);
  // 5) chunked attention
  attn_chunked_wmma<<<dim3(S_ / 64, NH_, B_), 256, 0, stream>>>(qb, kbuf, vbuf, ab);
  // 6) output projection -> f32 d_out
  gemm_bf16_wmma<1><<<dim3(M_ / 128, H_ / 64), 256, 0, stream>>>(ab, wo, d_out, M_, H_, H_);
}
